// K_Half_57758720197264
// MI455X (gfx1250) — matrix operand order repaired and checked by reading, not compiled, not run on hardware
//
#include <hip/hip_runtime.h>
#include <hip/hip_bf16.h>
#include <math.h>

// ---------------------------------------------------------------------------
// KBAT 2-layer GAT for MI455X (gfx1250, wave32).
// Dense projections: f16-staged WMMA GEMMs (padded operands -> branchless
// b128 fragment loads). Edge passes: float4 gather + f32 atomic scatter.
// ---------------------------------------------------------------------------

typedef __attribute__((ext_vector_type(16))) _Float16 v16h;
typedef __attribute__((ext_vector_type(8)))  _Float16 v8h;
typedef __attribute__((ext_vector_type(8)))  float    v8f;

#define ALPHA 0.2f
#define NRELS_C 200

// ---------------- pack A: f32 [rows,F] (ld=ldx) -> f16 [rowsPad,Fpad] -------
__global__ void k_pack_a(const float* __restrict__ X, int rows, int F, int ldx,
                         _Float16* __restrict__ out, int rowsPad, int Fpad)
{
    size_t idx = (size_t)blockIdx.x * blockDim.x + threadIdx.x;
    size_t total = (size_t)rowsPad * Fpad;
    if (idx >= total) return;
    int r = (int)(idx / Fpad);
    int f = (int)(idx % Fpad);
    float v = (r < rows && f < F) ? X[(size_t)r * ldx + f] : 0.0f;
    out[idx] = (_Float16)v;
}

// ---------------- pack B: W[c,f] = Wsrc[c*sC + f*sF] -> f16 [Cpad,Fpad] -----
__global__ void k_pack_b(const float* __restrict__ W, int C, int F, int sC, int sF,
                         _Float16* __restrict__ out, int Cpad, int Fpad)
{
    size_t idx = (size_t)blockIdx.x * blockDim.x + threadIdx.x;
    size_t total = (size_t)Cpad * Fpad;
    if (idx >= total) return;
    int c = (int)(idx / Fpad);
    int f = (int)(idx % Fpad);
    float v = (c < C && f < F) ? W[(size_t)c * sC + (size_t)f * sF] : 0.0f;
    out[idx] = (_Float16)v;
}

// ---------------- WMMA GEMM on padded f16 operands --------------------------
// out[m,n] = sum_k Xp[m,k] * Wp[n,k]  (both padded, Fpad % 32 == 0)
// A 16x32 f16 lane map (ISA 7.12.2): M=lane&15, K(i) = (i<8?i:8+i)+(hi?8:0)
//   -> per lane two contiguous runs of 8 halfs: [off, off+8) and [off+16, off+24)
// B 32x16 f16: N=lane&15, K(i) = same as A -> two contiguous runs of 8 halfs
// D 16x16 f32: N=lane&15, M(r) = r + (hi?8:0)
__global__ void k_gemm_wmma(const _Float16* __restrict__ Xp, int Fpad,
                            const _Float16* __restrict__ Wp,
                            int rows, int C,
                            float* __restrict__ out, int ldo)
{
    const int lane = threadIdx.x & 31;
    const int wave = (blockIdx.x * blockDim.x + threadIdx.x) >> 5;
    const int coltiles = (C + 15) >> 4;
    const int rowtiles = (rows + 15) >> 4;
    if (wave >= rowtiles * coltiles) return;          // wave-uniform exit
    const int rt = wave / coltiles;
    const int ct = wave % coltiles;
    const int l15 = lane & 15;
    const int hi  = (lane >> 4) & 1;

    const int m = rt * 16 + l15;
    const int n = ct * 16 + l15;
    const _Float16* arow = Xp + (size_t)m * Fpad + (hi ? 8 : 0);
    const _Float16* brow = Wp + (size_t)n * Fpad + (hi ? 8 : 0);

    v8f acc = {};
    const int ktiles = Fpad >> 5;
    for (int k0 = 0; k0 < ktiles; ++k0) {
        v8h a0 = *(const v8h*)(arow);
        v8h a1 = *(const v8h*)(arow + 16);
        v8h b0 = *(const v8h*)(brow);
        v8h b1 = *(const v8h*)(brow + 16);
        v16h av, bv;
#pragma unroll
        for (int i = 0; i < 8; ++i) {
            av[i] = a0[i]; av[i + 8] = a1[i];
            bv[i] = b0[i]; bv[i + 8] = b1[i];
        }
        acc = __builtin_amdgcn_wmma_f32_16x16x32_f16(
            false, av, false, bv, (short)0, acc, false, false);
        arow += 32;
        brow += 32;
    }
    const int mbase = rt * 16 + (hi ? 8 : 0);
    const bool nok = (n < C);
#pragma unroll
    for (int r = 0; r < 8; ++r) {
        int mm = mbase + r;
        if (mm < rows && nok) out[(size_t)mm * ldo + n] = acc[r];
    }
}

// ---------------- row dot with a2: out[r] = X[r,:] . a2 ---------------------
__global__ void k_dot_rows(const float* __restrict__ X, int rows, int F,
                           const float* __restrict__ a2, float* __restrict__ out)
{
    int r = blockIdx.x * blockDim.x + threadIdx.x;
    if (r >= rows) return;
    const float* x = X + (size_t)r * F;
    float s = 0.f;
    for (int f = 0; f < F; ++f) s += x[f] * a2[f];
    out[r] = s;
}

// ---------------- row l2 normalize -----------------------------------------
__global__ void k_l2norm_rows(const float* __restrict__ X, int rows, int F,
                              float* __restrict__ out)
{
    int r = blockIdx.x * blockDim.x + threadIdx.x;
    if (r >= rows) return;
    const float* x = X + (size_t)r * F;
    float ss = 0.f;
    for (int f = 0; f < F; ++f) ss += x[f] * x[f];
    float inv = 1.f / fmaxf(sqrtf(ss), 1e-12f);
    float* o = out + (size_t)r * F;
    for (int f = 0; f < F; ++f) o[f] = x[f] * inv;
}

// ---------------- edge scatter pass (wave per edge, grid-strided) -----------
// Accumulates e * (Xdst[e1] + Relm[t]) into Hacc[e0], and e into S[e0].
__global__ void k_edges(int E, const int* __restrict__ e0a, const int* __restrict__ e1a,
                        const int* __restrict__ eta, int dbl,
                        int nheads, int dim, int N,
                        const float* __restrict__ Xdst,   // [h][N][dim]
                        const float* __restrict__ Relm,   // [h][NRELS][dim]
                        const float* __restrict__ ssrc,   // [h][N]
                        const float* __restrict__ sdst,   // [h][N]
                        const float* __restrict__ srel,   // [h][NRELS]
                        float* __restrict__ Hacc,         // [N][nheads*dim]
                        float* __restrict__ S)            // [h][N]
{
    const int lane = threadIdx.x & 31;
    const int wid  = (blockIdx.x * blockDim.x + threadIdx.x) >> 5;
    const int nw   = (gridDim.x * blockDim.x) >> 5;
    const int q4   = dim >> 2;                        // float4s per row
    for (int e = wid; e < E; e += nw) {
        const int a = e0a[e];
        const int b = e1a[e];
        int t = eta[e];
        if (dbl) t = eta[t];
        // software prefetch of the next gathered row (global_prefetch_b8)
        if (lane == 0) {
            int en = e + nw;
            if (en < E) {
                int bn = e1a[en];
                __builtin_prefetch(Xdst + (size_t)bn * dim, 0, 0);
            }
        }
        for (int h = 0; h < nheads; ++h) {
            float s  = ssrc[(size_t)h * N + a] + sdst[(size_t)h * N + b]
                     + srel[h * NRELS_C + t];
            float lr = (s > 0.f) ? s : ALPHA * s;     // leaky_relu
            float ee = expf(-lr);                     // exp(powers)
            const float* xd = Xdst + ((size_t)h * N + b) * dim;
            const float* rm = Relm + ((size_t)h * NRELS_C + t) * dim;
            float* hp = Hacc + (size_t)a * (nheads * dim) + h * dim;
            for (int j = lane; j < q4; j += 32) {
                float4 x4 = *(const float4*)(xd + 4 * j);
                float4 r4 = *(const float4*)(rm + 4 * j);
                float* hp4 = hp + 4 * j;
                atomicAdd(hp4 + 0, ee * (x4.x + r4.x));
                atomicAdd(hp4 + 1, ee * (x4.y + r4.y));
                atomicAdd(hp4 + 2, ee * (x4.z + r4.z));
                atomicAdd(hp4 + 3, ee * (x4.w + r4.w));
            }
            if (lane == 0)
                atomicAdd(S + (size_t)h * N + a, ee);
        }
    }
}

// ---------------- finalize: x = elu((Xsrc*S' + Hacc) / (S' + eps)) ----------
__global__ void k_finalize(int N, int nheads, int dim,
                           const float* __restrict__ Xsrc, // [h][N][dim]
                           const float* __restrict__ S,    // [h][N]
                           float* __restrict__ H)          // [N][nheads*dim] in-place
{
    size_t idx = (size_t)blockIdx.x * blockDim.x + threadIdx.x;
    size_t total = (size_t)N * nheads * dim;
    if (idx >= total) return;
    int cols = nheads * dim;
    int v = (int)(idx / cols);
    int c = (int)(idx % cols);
    int h = c / dim;
    int j = c % dim;
    float Sp = S[(size_t)h * N + v];
    float denom = Sp + 1e-12f;
    float val = (Xsrc[((size_t)h * N + v) * dim + j] * Sp + H[idx]) / denom;
    H[idx] = (val > 0.f) ? val : (expf(val) - 1.f);        // elu
}

// ---------------- mask scatter ----------------------------------------------
__global__ void k_mask_set(const int* __restrict__ bi, int B, float* __restrict__ mask)
{
    int i = blockIdx.x * blockDim.x + threadIdx.x;
    if (i >= B) return;
    mask[bi[i * 4 + 2]] = 1.0f;
}

// ---------------- final: d_out = l2norm(entup + mask*x2), in place ----------
__global__ void k_final_out(int N, int C, float* __restrict__ out,
                            const float* __restrict__ x2,
                            const float* __restrict__ mask)
{
    int v = blockIdx.x * blockDim.x + threadIdx.x;
    if (v >= N) return;
    float mk = mask[v];
    float* o = out + (size_t)v * C;
    const float* x = x2 + (size_t)v * C;
    float ss = 0.f;
    for (int c = 0; c < C; ++c) {
        float u = o[c] + mk * x[c];
        o[c] = u;
        ss += u * u;
    }
    float inv = 1.f / fmaxf(sqrtf(ss), 1e-12f);
    for (int c = 0; c < C; ++c) o[c] *= inv;
}

// ---------------- his_temp_embs: out[i,j,k] = cos(t2[i]*w[k]+b[k]) ----------
__global__ void k_his(const int* __restrict__ bi, const float* __restrict__ w,
                      const float* __restrict__ bb, int B, int HDIM,
                      float* __restrict__ out)
{
    size_t idx = (size_t)blockIdx.x * blockDim.x + threadIdx.x;
    size_t total = (size_t)B * B * HDIM;
    if (idx >= total) return;
    int i = (int)(idx / ((size_t)B * HDIM));
    int k = (int)(idx % HDIM);
    float t2 = (float)bi[i * 4 + 3];
    out[idx] = cosf(t2 * w[k] + bb[k]);
}

// ---------------------------------------------------------------------------
static inline int cdiv(long long a, long long b) { return (int)((a + b - 1) / b); }
static inline int pad16(int x) { return (x + 15) & ~15; }
static inline int pad32(int x) { return (x + 31) & ~31; }

extern "C" void kernel_launch(void* const* d_in, const int* in_sizes, int n_in,
                              void* d_out, int out_size, void* d_ws, size_t ws_size,
                              hipStream_t stream)
{
    const float* ent    = (const float*)d_in[0];   // [N,100]
    const float* reltab = (const float*)d_in[1];   // [200,100]
    const float* w_t2   = (const float*)d_in[2];   // [100]
    const float* b_t2   = (const float*)d_in[3];   // [100]
    const float* W_ent  = (const float*)d_in[4];   // [100,200]
    const float* a_h    = (const float*)d_in[5];   // [2,100,300]
    const float* a2_h   = (const float*)d_in[6];   // [2,100]
    const float* W_rel  = (const float*)d_in[7];   // [100,200]
    const float* a_out  = (const float*)d_in[8];   // [200,600]
    const float* a2_out = (const float*)d_in[9];   // [200]
    const int*   elist  = (const int*)d_in[10];    // [2,E]
    const int*   etype  = (const int*)d_in[11];    // [E]
    const int*   binp   = (const int*)d_in[12];    // [B,4]

    const int FIN = 100, NHID = 100, H = 2, HDIM = 100;
    const int N  = in_sizes[0] / FIN;              // 50000 (multiple of 16)
    const int E  = in_sizes[11];                   // 800000
    const int B  = in_sizes[12] / 4;               // 512
    const int C2 = H * NHID;                       // 200

    const int FpF  = pad32(FIN);                   // 128
    const int FpC2 = pad32(C2);                    // 224
    const int NP   = pad16(N);                     // 50000
    const int RP   = pad16(NRELS_C);               // 208
    const int CpH  = pad16(NHID);                  // 112
    const int CpC2 = pad16(C2);                    // 208

    const int* edge0 = elist;
    const int* edge1 = elist + E;

    // ---- workspace layout ----
    char* wsb = (char*)d_ws;
    size_t off = 0;
    auto allocF = [&](size_t n) { float* p = (float*)(wsb + off); off += n * 4; return p; };
    auto allocH = [&](size_t n) { _Float16* p = (_Float16*)(wsb + off); off += n * 2;
                                  off = (off + 15) & ~(size_t)15; return p; };
    float* XS    = allocF((size_t)H * N * NHID);   // Xsrc L1 / X1src [N,200]
    float* XD    = allocF((size_t)H * N * NHID);   // Xdst L1 / X1dst
    float* X1    = allocF((size_t)N * C2);         // Hacc1 -> x1
    float* X2    = allocF((size_t)N * C2);         // Hacc2 -> x2
    float* EN    = allocF((size_t)N * FIN);        // emb_norm
    float* RELM1 = allocF((size_t)H * NRELS_C * NHID);
    float* R2TMP = allocF((size_t)NRELS_C * C2);
    float* RELM2 = allocF((size_t)NRELS_C * C2);
    float* SSRC  = allocF((size_t)H * N);
    float* SDST  = allocF((size_t)H * N);
    float* SREL  = allocF((size_t)H * NRELS_C);
    float* S1    = allocF((size_t)H * N);
    float* S2    = allocF((size_t)N);
    float* MASK  = allocF((size_t)N);
    // f16 staging
    _Float16* ENT16   = allocH((size_t)NP * FpF);
    _Float16* RT16    = allocH((size_t)RP * FpF);
    _Float16* X1P     = allocH((size_t)NP * FpC2);
    _Float16* R2T16   = allocH((size_t)RP * FpC2);
    _Float16* EN16    = allocH((size_t)NP * FpF);
    _Float16* AH16    = allocH((size_t)H * 3 * CpH * FpF);   // [h][slice][112][128]
    _Float16* AO16    = allocH((size_t)3 * CpC2 * FpC2);     // [slice][208][224]
    _Float16* WREL16  = allocH((size_t)CpC2 * FpF);
    _Float16* WENT16  = allocH((size_t)CpC2 * FpF);
    (void)ws_size;

    float* outp = (float*)d_out;                   // [N,200] then [B,B,100]
    float* hisp = outp + (size_t)N * C2;

    // ---- zero accumulators (graph-capture safe) ----
    hipMemsetAsync(X1,  0, (size_t)N * C2 * sizeof(float), stream);
    hipMemsetAsync(X2,  0, (size_t)N * C2 * sizeof(float), stream);
    hipMemsetAsync(S1,  0, (size_t)H * N * sizeof(float), stream);
    hipMemsetAsync(S2,  0, (size_t)N * sizeof(float), stream);
    hipMemsetAsync(MASK,0, (size_t)N * sizeof(float), stream);

    const int TB = 256;
    auto packs = [&](long long n) { return cdiv(n, TB); };
    auto gemm_blocks = [&](int rows, int C) {
        long long tiles = (long long)cdiv(rows, 16) * cdiv(C, 16);
        return cdiv(tiles, 8);                     // 8 waves / 256-thread block
    };

    // ---- stage constant operands to padded f16 ----
    k_pack_a<<<packs((long long)NP * FpF), TB, 0, stream>>>(ent, N, FIN, FIN, ENT16, NP, FpF);
    k_pack_a<<<packs((long long)RP * FpF), TB, 0, stream>>>(reltab, NRELS_C, FIN, FIN, RT16, RP, FpF);
    for (int h = 0; h < H; ++h)
        for (int s = 0; s < 3; ++s)
            k_pack_b<<<packs((long long)CpH * FpF), TB, 0, stream>>>(
                a_h + (size_t)h * NHID * 300 + s * 100, NHID, FIN, 300, 1,
                AH16 + ((size_t)h * 3 + s) * CpH * FpF, CpH, FpF);
    for (int s = 0; s < 3; ++s)
        k_pack_b<<<packs((long long)CpC2 * FpC2), TB, 0, stream>>>(
            a_out + s * C2, C2, C2, 3 * C2, 1, AO16 + (size_t)s * CpC2 * FpC2, CpC2, FpC2);
    k_pack_b<<<packs((long long)CpC2 * FpF), TB, 0, stream>>>(
        W_rel, C2, FIN, 1, C2, WREL16, CpC2, FpF);
    k_pack_b<<<packs((long long)CpC2 * FpF), TB, 0, stream>>>(
        W_ent, C2, FIN, 1, C2, WENT16, CpC2, FpF);

    // ---- emb_norm = l2norm(ent); stage it too ----
    k_l2norm_rows<<<cdiv(N, TB), TB, 0, stream>>>(ent, N, FIN, EN);
    k_pack_a<<<packs((long long)NP * FpF), TB, 0, stream>>>(EN, N, FIN, FIN, EN16, NP, FpF);

    // ---- layer-1 per-head projections via WMMA ----
    for (int h = 0; h < H; ++h) {
        const _Float16* asrc = AH16 + ((size_t)h * 3 + 0) * CpH * FpF;
        const _Float16* adst = AH16 + ((size_t)h * 3 + 1) * CpH * FpF;
        const _Float16* arel = AH16 + ((size_t)h * 3 + 2) * CpH * FpF;
        k_gemm_wmma<<<gemm_blocks(N, NHID), TB, 0, stream>>>(
            ENT16, FpF, asrc, N, NHID, XS + (size_t)h * N * NHID, NHID);
        k_gemm_wmma<<<gemm_blocks(N, NHID), TB, 0, stream>>>(
            ENT16, FpF, adst, N, NHID, XD + (size_t)h * N * NHID, NHID);
        k_gemm_wmma<<<gemm_blocks(NRELS_C, NHID), TB, 0, stream>>>(
            RT16, FpF, arel, NRELS_C, NHID, RELM1 + (size_t)h * NRELS_C * NHID, NHID);
        k_dot_rows<<<cdiv(N, TB), TB, 0, stream>>>(
            XS + (size_t)h * N * NHID, N, NHID, a2_h + h * NHID, SSRC + (size_t)h * N);
        k_dot_rows<<<cdiv(N, TB), TB, 0, stream>>>(
            XD + (size_t)h * N * NHID, N, NHID, a2_h + h * NHID, SDST + (size_t)h * N);
        k_dot_rows<<<cdiv(NRELS_C, TB), TB, 0, stream>>>(
            RELM1 + (size_t)h * NRELS_C * NHID, NRELS_C, NHID, a2_h + h * NHID,
            SREL + (size_t)h * NRELS_C);
    }

    // ---- layer-1 edge scatter + finalize (x1 in X1) ----
    k_edges<<<16384, TB, 0, stream>>>(E, edge0, edge1, etype, 0, H, NHID, N,
                                      XD, RELM1, SSRC, SDST, SREL, X1, S1);
    k_finalize<<<cdiv((long long)N * C2, TB), TB, 0, stream>>>(N, H, NHID, XS, S1, X1);

    // ---- layer-2 projections (reuse XS/XD) ----
    k_pack_a<<<packs((long long)NP * FpC2), TB, 0, stream>>>(X1, N, C2, C2, X1P, NP, FpC2);
    k_gemm_wmma<<<gemm_blocks(N, C2), TB, 0, stream>>>(
        X1P, FpC2, AO16 + (size_t)0 * CpC2 * FpC2, N, C2, XS, C2);
    k_gemm_wmma<<<gemm_blocks(N, C2), TB, 0, stream>>>(
        X1P, FpC2, AO16 + (size_t)1 * CpC2 * FpC2, N, C2, XD, C2);
    k_gemm_wmma<<<gemm_blocks(NRELS_C, C2), TB, 0, stream>>>(
        RT16, FpF, WREL16, NRELS_C, C2, R2TMP, C2);             // reltab @ W_rel
    k_pack_a<<<packs((long long)RP * FpC2), TB, 0, stream>>>(R2TMP, NRELS_C, C2, C2, R2T16, RP, FpC2);
    k_gemm_wmma<<<gemm_blocks(NRELS_C, C2), TB, 0, stream>>>(
        R2T16, FpC2, AO16 + (size_t)2 * CpC2 * FpC2, NRELS_C, C2, RELM2, C2);
    k_dot_rows<<<cdiv(N, TB), TB, 0, stream>>>(XS, N, C2, a2_out, SSRC);
    k_dot_rows<<<cdiv(N, TB), TB, 0, stream>>>(XD, N, C2, a2_out, SDST);
    k_dot_rows<<<cdiv(NRELS_C, TB), TB, 0, stream>>>(RELM2, NRELS_C, C2, a2_out, SREL);

    // ---- layer-2 edge scatter (double-indexed edge_type quirk) + finalize ----
    k_edges<<<16384, TB, 0, stream>>>(E, edge0, edge1, etype, 1, 1, C2, N,
                                      XD, RELM2, SSRC, SDST, SREL, X2, S2);
    k_finalize<<<cdiv((long long)N * C2, TB), TB, 0, stream>>>(N, 1, C2, XS, S2, X2);

    // ---- entities_upgraded = emb_norm @ W_entities -> d_out, then mask+l2norm ----
    k_gemm_wmma<<<gemm_blocks(N, C2), TB, 0, stream>>>(
        EN16, FpF, WENT16, N, C2, outp, C2);
    k_mask_set<<<cdiv(B, TB), TB, 0, stream>>>(binp, B, MASK);
    k_final_out<<<cdiv(N, TB), TB, 0, stream>>>(N, C2, outp, X2, MASK);

    // ---- his_temp_embs broadcast ----
    k_his<<<cdiv((long long)B * B * HDIM, TB), TB, 0, stream>>>(
        binp, w_t2, b_t2, B, HDIM, hisp);
}
